// ColumnCompute_11055245820017
// MI455X (gfx1250) — matrix operand order repaired and checked by reading, not compiled, not run on hardware
//
#include <hip/hip_runtime.h>
#include <hip/hip_bf16.h>

// ---------------- types ----------------
typedef __attribute__((ext_vector_type(16))) _Float16 v16h;
typedef __attribute__((ext_vector_type(8)))  _Float16 v8h;
typedef __attribute__((ext_vector_type(4)))  _Float16 v4h;
typedef __attribute__((ext_vector_type(8)))  float    v8f;

#define BM 128
#define BN 128
#define BK 64
#define LDA_P 72   // padded halfs per A-tile row (144 B pitch, 16B-aligned)
#define LDB_P 72   // padded halfs per B-tile (transposed) row

// Problem constants (match reference)
static constexpr int   Bb     = 1024;
static constexpr int   Hh     = 32;
static constexpr int   Ll     = 3;
static constexpr int   DS     = 512;
static constexpr int   DID    = 128;
static constexpr int   DSTEER = 1664;
static constexpr int   DHID   = 2048;
static constexpr int   HDQ_   = 1024;
static constexpr int   BH     = Bb * Hh;       // 32768
static constexpr long long OUTW = 2560;        // DS + HDQ + HDQ

struct GemmP {
  const float* __restrict__ A;    long long lda,  a_batch;
  const float* __restrict__ Bm;   long long ldb,  b_batch;
  const float* __restrict__ bias; long long bias_batch;
  const float* __restrict__ res;  long long ldr,  res_batch;
  float*       __restrict__ C;    long long ldc,  c_batch;
  int M, N, K;
  int act;   // 0 = none, 1 = exact GELU
};

__device__ __forceinline__ float gelu_exact(float x) {
  return 0.5f * x * (1.0f + erff(x * 0.7071067811865475f));
}

union V16U { v16h v; v8h h[2]; };

// ---- tile stage helpers (register prefetch -> LDS) ----
__device__ __forceinline__ void load_a_regs(const float* __restrict__ A, long long lda,
                                            int row0, int k0, int tid, float4 (&ar)[8]) {
  #pragma unroll
  for (int j = 0; j < 8; ++j) {
    int idx = tid + j * 256;       // 0..2047 over BM x (BK/4)
    int r   = idx >> 4;            // 0..127
    int c   = (idx & 15) << 2;     // 0..60 step 4
    ar[j] = *(const float4*)(A + (long long)(row0 + r) * lda + (k0 + c));
  }
}

__device__ __forceinline__ void store_a_lds(_Float16* As, int tid, const float4 (&ar)[8]) {
  #pragma unroll
  for (int j = 0; j < 8; ++j) {
    int idx = tid + j * 256;
    int r   = idx >> 4;
    int c   = (idx & 15) << 2;
    v4h hv = { (_Float16)ar[j].x, (_Float16)ar[j].y, (_Float16)ar[j].z, (_Float16)ar[j].w };
    *(v4h*)&As[r * LDA_P + c] = hv;   // one ds_store_b64 per 4 elements
  }
}

// B loaded column-wise: 4 consecutive K at fixed N. Each b32 global load is
// coalesced across lanes (lanes cover consecutive N); LDS store is one b64.
__device__ __forceinline__ void load_b_regs(const float* __restrict__ B, long long ldb,
                                            int col0, int k0, int tid, float4 (&br)[8]) {
  #pragma unroll
  for (int j = 0; j < 8; ++j) {
    int idx = tid + j * 256;        // 0..2047 over (BK/4) x BN
    int n   = idx & 127;            // 0..127
    int kg  = (idx >> 7) << 2;      // 0,4,...,60
    const float* gp = B + (long long)(k0 + kg) * ldb + (col0 + n);
    br[j].x = gp[0];
    br[j].y = gp[ldb];
    br[j].z = gp[2 * ldb];
    br[j].w = gp[3 * ldb];
  }
}

__device__ __forceinline__ void store_b_lds(_Float16* Bs, int tid, const float4 (&br)[8]) {
  #pragma unroll
  for (int j = 0; j < 8; ++j) {
    int idx = tid + j * 256;
    int n   = idx & 127;
    int kg  = (idx >> 7) << 2;
    v4h hv = { (_Float16)br[j].x, (_Float16)br[j].y, (_Float16)br[j].z, (_Float16)br[j].w };
    *(v4h*)&Bs[n * LDB_P + kg] = hv;  // transposed [n][k], one ds_store_b64
  }
}

// Batched GEMM: C[M,N] = act(A[M,K] @ B[K,N] + bias) + res
// fp32 global operands, fp32->f16 conversion fused into the tile stage,
// v_wmma_f32_16x16x32_f16 accumulation, register-stage prefetch pipeline.
__global__ __launch_bounds__(256) void wmma_gemm_f16(GemmP p) {
  __shared__ _Float16 As[BM * LDA_P];
  __shared__ _Float16 Bs[BN * LDB_P];

  const int z = blockIdx.z;
  const float* A    = p.A  + (long long)z * p.a_batch;
  const float* B    = p.Bm + (long long)z * p.b_batch;
  const float* bias = p.bias ? p.bias + (long long)z * p.bias_batch : nullptr;
  const float* res  = p.res  ? p.res  + (long long)z * p.res_batch  : nullptr;
  float*       C    = p.C  + (long long)z * p.c_batch;

  const int row0 = blockIdx.y * BM;
  const int col0 = blockIdx.x * BN;

  const int tid  = threadIdx.x;
  const int lane = tid & 31;
  const int wave = tid >> 5;
  const int wm   = (wave & 1) * 64;   // wave's M offset within block tile
  const int wn   = (wave >> 1) * 32;  // wave's N offset within block tile
  const int lrow = lane & 15;
  const int lhi  = lane >> 4;

  v8f acc[4][2];
  v8f zv = {0.f, 0.f, 0.f, 0.f, 0.f, 0.f, 0.f, 0.f};
  #pragma unroll
  for (int t = 0; t < 4; ++t)
    #pragma unroll
    for (int u = 0; u < 2; ++u)
      acc[t][u] = zv;

  float4 ar[8], br[8];
  load_a_regs(A, p.lda, row0, 0, tid, ar);
  load_b_regs(B, p.ldb, col0, 0, tid, br);

  for (int k0 = 0; k0 < p.K; k0 += BK) {
    store_a_lds(As, tid, ar);
    store_b_lds(Bs, tid, br);
    __syncthreads();

    // prefetch next tile into registers while WMMAs consume LDS
    if (k0 + BK < p.K) {
      load_a_regs(A, p.lda, row0, k0 + BK, tid, ar);
      load_b_regs(B, p.ldb, col0, k0 + BK, tid, br);
    }

    #pragma unroll
    for (int ph = 0; ph < 2; ++ph) {        // two 32-deep WMMA phases per tile
      V16U a[4], b[2];
      #pragma unroll
      for (int t = 0; t < 4; ++t) {
        int r  = wm + t * 16 + lrow;
        int kh = ph * 32 + lhi * 8;
        a[t].h[0] = *(const v8h*)&As[r * LDA_P + kh];
        a[t].h[1] = *(const v8h*)&As[r * LDA_P + kh + 16];
      }
      #pragma unroll
      for (int u = 0; u < 2; ++u) {
        int cc = wn + u * 16 + lrow;
        int kb = ph * 32 + lhi * 8;
        b[u].h[0] = *(const v8h*)&Bs[cc * LDB_P + kb];
        b[u].h[1] = *(const v8h*)&Bs[cc * LDB_P + kb + 16];
      }
      #pragma unroll
      for (int t = 0; t < 4; ++t)
        #pragma unroll
        for (int u = 0; u < 2; ++u)
          acc[t][u] = __builtin_amdgcn_wmma_f32_16x16x32_f16(
              false, a[t].v, false, b[u].v, (short)0, acc[t][u], false, false);
    }
    __syncthreads();
  }

  // ---- epilogue: bias -> activation -> residual -> fp32 store ----
  #pragma unroll
  for (int t = 0; t < 4; ++t) {
    #pragma unroll
    for (int u = 0; u < 2; ++u) {
      int n = col0 + wn + u * 16 + lrow;
      float bv = bias ? bias[n] : 0.0f;
      #pragma unroll
      for (int v = 0; v < 8; ++v) {
        long long r = (long long)(row0 + wm + t * 16 + v + 8 * lhi);
        float xv = acc[t][u][v] + bv;
        if (p.act == 1) xv = gelu_exact(xv);
        if (res) xv += res[r * p.ldr + n];
        C[r * p.ldc + n] = xv;
      }
    }
  }
}

// Per-row RMSNorm with per-head gain: xn = h * rsqrt(mean(h^2)+1e-6) * w[head]
__global__ __launch_bounds__(256) void rmsnorm_kernel(
    const float* __restrict__ h, const float* __restrict__ nw,
    float* __restrict__ xn, int D, int nheads, long long nw_head_stride) {
  int row  = blockIdx.x;
  int head = row % nheads;
  const float* hr = h + (long long)row * D;
  float s = 0.f;
  for (int d = threadIdx.x; d < D; d += 256) { float v = hr[d]; s += v * v; }
  __shared__ float red[256];
  red[threadIdx.x] = s;
  __syncthreads();
  for (int off = 128; off > 0; off >>= 1) {
    if (threadIdx.x < off) red[threadIdx.x] += red[threadIdx.x + off];
    __syncthreads();
  }
  float rms = rsqrtf(red[0] / (float)D + 1e-6f);
  const float* w = nw + (long long)head * nw_head_stride;
  for (int d = threadIdx.x; d < D; d += 256)
    xn[(long long)row * D + d] = hr[d] * rms * w[d];
}

// Copy dense [rows, D] into strided output columns.
__global__ void copy_to_out(const float* __restrict__ src, float* __restrict__ dst,
                            int D, long long ldd, long long total) {
  long long i = (long long)blockIdx.x * blockDim.x + threadIdx.x;
  if (i >= total) return;
  long long r = i / D;
  int d = (int)(i - r * D);
  dst[r * ldd + d] = src[i];
}

extern "C" void kernel_launch(void* const* d_in, const int* in_sizes, int n_in,
                              void* d_out, int out_size, void* d_ws, size_t ws_size,
                              hipStream_t stream) {
  (void)in_sizes; (void)n_in; (void)out_size; (void)ws_size;
  const float* x         = (const float*)d_in[0];
  const float* id_in     = (const float*)d_in[1];
  const float* in_proj_w = (const float*)d_in[2];
  const float* in_proj_b = (const float*)d_in[3];
  const float* norm_w    = (const float*)d_in[4];
  const float* up_w      = (const float*)d_in[5];
  const float* up_b      = (const float*)d_in[6];
  const float* down_w    = (const float*)d_in[7];
  const float* down_b    = (const float*)d_in[8];
  const float* q1_w      = (const float*)d_in[9];
  const float* q1_b      = (const float*)d_in[10];
  const float* q2_w      = (const float*)d_in[11];
  const float* q2_b      = (const float*)d_in[12];
  const float* k1_w      = (const float*)d_in[13];
  const float* k1_b      = (const float*)d_in[14];
  const float* k2_w      = (const float*)d_in[15];
  const float* k2_b      = (const float*)d_in[16];
  float* out = (float*)d_out;

  // workspace: h[BH,DS] | xn[BH,DS] | t[BH,DHID] (t reused by hid, q-mid, k-mid)
  float* h  = (float*)d_ws;
  float* xn = h  + (long long)BH * DS;
  float* t  = xn + (long long)BH * DS;

  auto launch = [&](const float* A, long long lda, long long ab,
                    const float* Bm, long long ldb, long long bb,
                    const float* bias, long long biasb,
                    const float* res, long long ldr, long long resb,
                    float* C, long long ldc, long long cb,
                    int M, int N, int K, int act, int batches) {
    GemmP p{A, lda, ab, Bm, ldb, bb, bias, biasb, res, ldr, resb, C, ldc, cb, M, N, K, act};
    dim3 grid((unsigned)(N / BN), (unsigned)(M / BM), (unsigned)batches);
    wmma_gemm_f16<<<grid, dim3(256), 0, stream>>>(p);
  };

  // ---- content path: per-head in_proj (batched over 32 heads) ----
  launch(x, (long long)Hh * DSTEER, DSTEER,
         in_proj_w, DS, (long long)DSTEER * DS,
         in_proj_b, DS,
         nullptr, 0, 0,
         h, (long long)Hh * DS, DS,
         Bb, DS, DSTEER, /*act*/0, Hh);

  for (int i = 0; i < Ll; ++i) {
    rmsnorm_kernel<<<dim3(BH), dim3(256), 0, stream>>>(
        h, norm_w + (long long)i * DS, xn, DS, Hh, (long long)Ll * DS);

    // up: gelu(xn @ up_w[:,i] + up_b[:,i]) -> t (hid)
    launch(xn, (long long)Hh * DS, DS,
           up_w + (long long)i * DS * DHID, DHID, (long long)Ll * DS * DHID,
           up_b + (long long)i * DHID, (long long)Ll * DHID,
           nullptr, 0, 0,
           t, (long long)Hh * DHID, DHID,
           Bb, DHID, DS, /*act*/1, Hh);

    // down: h += hid @ down_w[:,i] + down_b[:,i]  (residual fused, in-place)
    launch(t, (long long)Hh * DHID, DHID,
           down_w + (long long)i * DHID * DS, DS, (long long)Ll * DHID * DS,
           down_b + (long long)i * DS, (long long)Ll * DS,
           h, (long long)Hh * DS, DS,
           h, (long long)Hh * DS, DS,
           Bb, DS, DHID, /*act*/0, Hh);
  }

  // content -> out[:, 0:512]
  {
    long long total = (long long)BH * DS;
    unsigned blocks = (unsigned)((total + 255) / 256);
    copy_to_out<<<dim3(blocks), dim3(256), 0, stream>>>(h, out, DS, OUTW, total);
  }

  // ---- q path ----
  launch(x, DSTEER, 0, q1_w, 2 * HDQ_, 0, q1_b, 0, nullptr, 0, 0,
         t, 2 * HDQ_, 0, BH, 2 * HDQ_, DSTEER, /*act*/1, 1);
  launch(t, 2 * HDQ_, 0, q2_w, HDQ_, 0, q2_b, 0, nullptr, 0, 0,
         out + DS, OUTW, 0, BH, HDQ_, 2 * HDQ_, /*act*/0, 1);

  // ---- k path ----
  launch(id_in, DID, 0, k1_w, 2 * HDQ_, 0, k1_b, 0, nullptr, 0, 0,
         t, 2 * HDQ_, 0, BH, 2 * HDQ_, DID, /*act*/1, 1);
  launch(t, 2 * HDQ_, 0, k2_w, HDQ_, 0, k2_b, 0, nullptr, 0, 0,
         out + DS + HDQ_, OUTW, 0, BH, HDQ_, 2 * HDQ_, /*act*/0, 1);
}
